// DenseContrastiveLoss_48155173323219
// MI455X (gfx1250) — matrix operand order repaired and checked by reading, not compiled, not run on hardware
//
#include <hip/hip_runtime.h>

// ---------------------------------------------------------------------------
// DenseContrastiveLoss on MI455X (gfx1250, wave32)
//   X = concat(f1,f2) : (512, 65536) fp32
//   gram = X X^T via split-f16 WMMA (hi*hi + hi*lo + lo*hi), split-K=16,
//   partials accumulated with global f32 atomics into workspace.
//   Then faithful row-softmax contrastive loss -> scalar mean.
// ---------------------------------------------------------------------------

typedef __attribute__((ext_vector_type(16))) _Float16 v16h;
typedef __attribute__((ext_vector_type(8)))  float    v8f;

#define NROWS   512
#define KDIM    65536
#define TILE    128
#define KSTEP   32
#define KSPLIT  16
#define TEMP    0.01f
// LDS row: 32 f16 = 16 dwords data + 4 dwords pad = 20 dwords (80 B, 16B-aligned,
// 80 mod 256 visits all bank groups -> conflict-free b128 fragment reads)
#define ROWW    20
#define PLANEW  (TILE * ROWW)   // 2560 dwords per plane; 4 planes = 40 KB LDS

__device__ __forceinline__ const float* row_ptr(const float* f1, const float* f2, int r) {
    return (r < 256) ? (f1 + (size_t)r * KDIM) : (f2 + (size_t)(r - 256) * KDIM);
}

__device__ __forceinline__ v16h make_frag(uint4 a, uint4 b) {
    struct alignas(16) Pair { uint4 a; uint4 b; };
    Pair p; p.a = a; p.b = b;
    return __builtin_bit_cast(v16h, p);   // 32 bytes -> 16 x f16
}

// Convert 4 fp32 to (hi,lo) f16 pairs and store 2 dwords to each LDS plane.
__device__ __forceinline__ void store_quad(unsigned int* Ph, unsigned int* Pl,
                                           int row, int kq, float4 v) {
    float f[4] = { v.x, v.y, v.z, v.w };
    unsigned short hb[4], lb[4];
#pragma unroll
    for (int i = 0; i < 4; ++i) {
        _Float16 h = (_Float16)f[i];
        float hf   = (float)h;
        _Float16 l = (_Float16)(f[i] - hf);
        hb[i] = __builtin_bit_cast(unsigned short, h);
        lb[i] = __builtin_bit_cast(unsigned short, l);
    }
    const int w = row * ROWW + (kq >> 1);          // dword index (8B aligned)
    uint2 hv; hv.x = (unsigned)hb[0] | ((unsigned)hb[1] << 16);
              hv.y = (unsigned)hb[2] | ((unsigned)hb[3] << 16);
    uint2 lv; lv.x = (unsigned)lb[0] | ((unsigned)lb[1] << 16);
              lv.y = (unsigned)lb[2] | ((unsigned)lb[3] << 16);
    *(uint2*)(&Ph[w]) = hv;
    *(uint2*)(&Pl[w]) = lv;
}

// ---------------------------------------------------------------------------
// Kernel: zero gram workspace (poisoned by harness; atomics need zeros)
// ---------------------------------------------------------------------------
__global__ void zero_kernel(float* __restrict__ g) {
    int i = blockIdx.x * blockDim.x + threadIdx.x;   // 65536 threads * float4
    float4 z; z.x = z.y = z.z = z.w = 0.0f;
    ((float4*)g)[i] = z;
}

// ---------------------------------------------------------------------------
// Kernel: sq[i] = ||x_i||^2 (fp32, one block per row)
// ---------------------------------------------------------------------------
__global__ __launch_bounds__(256)
void sq_kernel(const float* __restrict__ f1, const float* __restrict__ f2,
               float* __restrict__ sq) {
    const int row = blockIdx.x;
    const float* p = row_ptr(f1, f2, row);
    float s = 0.0f;
    for (int i4 = threadIdx.x; i4 < KDIM / 4; i4 += 256) {
        float4 v = *(const float4*)(p + (size_t)i4 * 4);
        s += v.x * v.x + v.y * v.y + v.z * v.z + v.w * v.w;
    }
    __shared__ float red[256];
    red[threadIdx.x] = s; __syncthreads();
    for (int off = 128; off > 0; off >>= 1) {
        if (threadIdx.x < off) red[threadIdx.x] += red[threadIdx.x + off];
        __syncthreads();
    }
    if (threadIdx.x == 0) sq[row] = red[0];
}

// ---------------------------------------------------------------------------
// Kernel: gram += X_blk_i . X_blk_j^T over this WG's K-chunk (split-f16 WMMA)
// grid (16 output tiles, 16 K-chunks), 256 threads = 8 wave32.
// ---------------------------------------------------------------------------
__global__ __launch_bounds__(256)
void gram_kernel(const float* __restrict__ f1, const float* __restrict__ f2,
                 float* __restrict__ gram) {
    __shared__ __align__(16) unsigned int lds[4 * PLANEW];
    unsigned int* Ah = lds;
    unsigned int* Al = lds + PLANEW;
    unsigned int* Bh = lds + 2 * PLANEW;
    unsigned int* Bl = lds + 3 * PLANEW;

    const int bi    = blockIdx.x >> 2;
    const int bj    = blockIdx.x & 3;
    const int kc    = blockIdx.y;
    const int tid   = threadIdx.x;
    const int lane  = tid & 31;
    const int wave  = tid >> 5;
    const int lrow  = lane & 15;
    const int lhalf = lane >> 4;
    const int wr    = wave >> 1;   // 0..3 : 32-row band of the 128x128 tile
    const int wc    = wave & 1;    // 0..1 : 64-col band

    v8f acc[2][4];
#pragma unroll
    for (int m = 0; m < 2; ++m)
#pragma unroll
        for (int n = 0; n < 4; ++n)
#pragma unroll
            for (int e = 0; e < 8; ++e) acc[m][n][e] = 0.0f;

    const int kbeg = kc * (KDIM / KSPLIT);
    const int kend = kbeg + (KDIM / KSPLIT);

    for (int k0 = kbeg; k0 < kend; k0 += KSTEP) {
        __syncthreads();                       // previous stage's reads done
        // --- stage: 128 rows x 32 k fp32 for each panel -> hi/lo f16 in LDS
#pragma unroll
        for (int it = 0; it < 4; ++it) {
            int q   = tid + it * 256;          // 0..1023 float4-quads
            int row = q >> 3;
            int kq  = (q & 7) * 4;
            {
                const float* p = row_ptr(f1, f2, bi * TILE + row) + k0 + kq;
                store_quad(Ah, Al, row, kq, *(const float4*)p);
            }
            {
                const float* p = row_ptr(f1, f2, bj * TILE + row) + k0 + kq;
                store_quad(Bh, Bl, row, kq, *(const float4*)p);
            }
        }
        __syncthreads();

        // --- A fragments: 16x32 f16; lanes 0-15 take K{0..7,16..23},
        //     lanes 16-31 take K{8..15,24..31}  (ISA 7.12.2) -> 2x ds_load_b128
        v16h ah[2], al[2];
#pragma unroll
        for (int m = 0; m < 2; ++m) {
            int r    = wr * 32 + m * 16 + lrow;
            int base = r * ROWW + (lhalf ? 4 : 0);
            ah[m] = make_frag(*(const uint4*)(&Ah[base]),
                              *(const uint4*)(&Ah[base + 8]));
            al[m] = make_frag(*(const uint4*)(&Al[base]),
                              *(const uint4*)(&Al[base + 8]));
        }
        // --- B fragments: 32x16; lane = column, lanes 0-15 K{0..7,16..23}, 16-31 K{8..15,24..31}
#pragma unroll
        for (int n = 0; n < 4; ++n) {
            int c    = wc * 64 + n * 16 + lrow;
            int base = c * ROWW + (lhalf ? 4 : 0);
            v16h bh = make_frag(*(const uint4*)(&Bh[base]),
                                *(const uint4*)(&Bh[base + 8]));
            v16h bl = make_frag(*(const uint4*)(&Bl[base]),
                                *(const uint4*)(&Bl[base + 8]));
#pragma unroll
            for (int m = 0; m < 2; ++m) {
                acc[m][n] = __builtin_amdgcn_wmma_f32_16x16x32_f16(
                    false, ah[m], false, bh, (short)0, acc[m][n], false, false);
                acc[m][n] = __builtin_amdgcn_wmma_f32_16x16x32_f16(
                    false, ah[m], false, bl, (short)0, acc[m][n], false, false);
                acc[m][n] = __builtin_amdgcn_wmma_f32_16x16x32_f16(
                    false, al[m], false, bh, (short)0, acc[m][n], false, false);
            }
        }
    }

    // --- writeback: C/D layout = VGPR r -> M=r (lanes<16) / M=8+r (lanes>=16)
#pragma unroll
    for (int m = 0; m < 2; ++m) {
        int row0 = bi * TILE + wr * 32 + m * 16 + lhalf * 8;
#pragma unroll
        for (int n = 0; n < 4; ++n) {
            int col = bj * TILE + wc * 64 + n * 16 + lrow;
#pragma unroll
            for (int r = 0; r < 8; ++r)
                atomicAdd(&gram[(size_t)(row0 + r) * NROWS + col], acc[m][n][r]);
        }
    }
}

// ---------------------------------------------------------------------------
// Kernel: per-row contrastive loss (faithful to the JAX reference)
// ---------------------------------------------------------------------------
__global__ __launch_bounds__(256)
void loss_kernel(const float* __restrict__ gram, const float* __restrict__ sq,
                 float* __restrict__ rloss) {
    const int i   = blockIdx.x;
    const int tid = threadIdx.x;
    const float sqi = sq[i];
    const float* g  = gram + (size_t)i * NROWS;

    float lg[2];
    float m = -3.402823e38f;
#pragma unroll
    for (int t = 0; t < 2; ++t) {
        int c = tid + t * 256;
        float l = -TEMP * (sqi + sq[c] - 2.0f * g[c]);   // includes diagonal
        lg[t] = l;
        m = fmaxf(m, l);
    }
    __shared__ float red[256];
    red[tid] = m; __syncthreads();
    for (int off = 128; off > 0; off >>= 1) {
        if (tid < off) red[tid] = fmaxf(red[tid], red[tid + off]);
        __syncthreads();
    }
    const float M = red[0];
    __syncthreads();

    float s = 0.0f;
#pragma unroll
    for (int t = 0; t < 2; ++t) {
        int c = tid + t * 256;
        if (c != i) s += expf(lg[t] - M);                // logits_mask = 1-eye
    }
    red[tid] = s; __syncthreads();
    for (int off = 128; off > 0; off >>= 1) {
        if (tid < off) red[tid] += red[tid + off];
        __syncthreads();
    }
    if (tid == 0) {
        int p = i ^ 256;                                 // unique positive pair
        float lpg = -TEMP * (sqi + sq[p] - 2.0f * g[p]);
        float logprob = (lpg - M) - logf(red[0] + 1e-10f);
        rloss[i] = -logprob;                             // TEMP/BASE_TEMP = 1
    }
}

__global__ __launch_bounds__(256)
void final_kernel(const float* __restrict__ rloss, float* __restrict__ out) {
    __shared__ float red[256];
    float s = rloss[threadIdx.x] + rloss[threadIdx.x + 256];
    red[threadIdx.x] = s; __syncthreads();
    for (int off = 128; off > 0; off >>= 1) {
        if (threadIdx.x < off) red[threadIdx.x] += red[threadIdx.x + off];
        __syncthreads();
    }
    if (threadIdx.x == 0) out[0] = red[0] * (1.0f / 512.0f);
}

// ---------------------------------------------------------------------------
extern "C" void kernel_launch(void* const* d_in, const int* in_sizes, int n_in,
                              void* d_out, int out_size, void* d_ws, size_t ws_size,
                              hipStream_t stream) {
    const float* f1 = (const float*)d_in[0];
    const float* f2 = (const float*)d_in[1];
    float* out   = (float*)d_out;
    float* ws    = (float*)d_ws;
    float* sq    = ws;                       // 512
    float* gram  = ws + 512;                 // 512*512
    float* rloss = ws + 512 + NROWS * NROWS; // 512   (~1.03 MB total)

    zero_kernel <<<256, 256, 0, stream>>>(gram);
    sq_kernel   <<<512, 256, 0, stream>>>(f1, f2, sq);
    gram_kernel <<<dim3(16, KSPLIT), 256, 0, stream>>>(f1, f2, gram);
    loss_kernel <<<512, 256, 0, stream>>>(gram, sq, rloss);
    final_kernel<<<1,   256, 0, stream>>>(rloss, out);
}
